// MultiHeadAttention_26319559590655
// MI455X (gfx1250) — compile-verified, matrix operand order checked by reading, not run on hardware
//
#include <hip/hip_runtime.h>

#define B_  4
#define S_  2048
#define D_  1024
#define H_  16
#define HD_ 64

typedef __attribute__((ext_vector_type(16))) __bf16 v16bf;
typedef __attribute__((ext_vector_type(8)))  float  v8f;

static __device__ __forceinline__ v8f v8f_zero() {
  v8f z;
#pragma unroll
  for (int i = 0; i < 8; ++i) z[i] = 0.0f;
  return z;
}

// fp32 -> bf16, round to nearest even
static __device__ __forceinline__ __bf16 f2bf(float f) {
  unsigned u = __builtin_bit_cast(unsigned, f);
  u += 0x7FFFu + ((u >> 16) & 1u);
  unsigned short h = (unsigned short)(u >> 16);
  return __builtin_bit_cast(__bf16, h);
}

static __device__ __forceinline__ v8f wmma_bf16(v16bf a, v16bf b, v8f c) {
  // (neg_a, A, neg_b, B, c_mod, C, reuse_a, reuse_b)
  return __builtin_amdgcn_wmma_f32_16x16x32_bf16(false, a, false, b, (short)0, c,
                                                 false, false);
}

// CDNA5 async global->LDS copy (16B per lane), tracked by ASYNCcnt.
// VDST = wave-relative LDS byte address (low 32 bits of a generic shared
// pointer, per the ISA flat->LDS truncation rule), VADDR = 64-bit global addr.
static __device__ __forceinline__ void async_copy_b128(void* lds_dst, const void* gsrc) {
  const unsigned ldsa = (unsigned)(uintptr_t)lds_dst;
  asm volatile("global_load_async_to_lds_b128 %0, %1, off"
               :: "v"(ldsa), "v"(gsrc)
               : "memory");
}

static __device__ __forceinline__ void wait_async0() {
#if __has_builtin(__builtin_amdgcn_s_wait_asynccnt)
  __builtin_amdgcn_s_wait_asynccnt(0);
#else
  asm volatile("s_wait_asynccnt 0x0" ::: "memory");
#endif
}

// 16x32 bf16 tile in WMMA A layout: lane holds row, K subset
// {kb..kb+7, kb+16..kb+23}, kb = 8*(lane>=16).  The B operand uses lane=column
// with the same K striping, so this loader also serves B tiles whenever the
// source is K-contiguous for a fixed column (Wt[N][K], the LDS K/V/P tiles).
static __device__ __forceinline__ v16bf load_tile_bf16(const __bf16* p, int ld,
                                                       int row, int kBase, int lane) {
  const int kb = kBase + ((lane >> 4) & 1) * 8;
  const __bf16* q = p + (size_t)row * ld + kb;
  v16bf a;
#pragma unroll
  for (int i = 0; i < 8; ++i) a[i] = q[i];
#pragma unroll
  for (int i = 0; i < 8; ++i) a[8 + i] = q[16 + i];
  return a;
}

static __device__ __forceinline__ v16bf load_tile_f32(const float* p, int ld,
                                                      int row, int kBase, int lane) {
  const int kb = kBase + ((lane >> 4) & 1) * 8;
  const float* q = p + (size_t)row * ld + kb;
  v16bf a;
#pragma unroll
  for (int i = 0; i < 8; ++i) a[i] = f2bf(q[i]);
#pragma unroll
  for (int i = 0; i < 8; ++i) a[8 + i] = f2bf(q[16 + i]);
  return a;
}

// One-time weight prep: W[K][N] fp32 -> Wt[N][K] bf16 via a 64x64 LDS tile.
__global__ __launch_bounds__(256) void transpose_w_bf16(const float* __restrict__ W,
                                                        __bf16* __restrict__ Wt) {
  __shared__ __align__(16) __bf16 tile[64][72];  // 144B row stride: 16B aligned
  const int k0 = blockIdx.x * 64;
  const int n0 = blockIdx.y * 64;
  const int t = threadIdx.x;
  const int tr = t >> 4;        // 0..15
  const int tc = (t & 15) * 4;  // 0..60
#pragma unroll
  for (int i = 0; i < 4; ++i) {
    const float4 v = *(const float4*)(W + (size_t)(k0 + tr + i * 16) * D_ + n0 + tc);
    tile[tc + 0][tr + i * 16] = f2bf(v.x);
    tile[tc + 1][tr + i * 16] = f2bf(v.y);
    tile[tc + 2][tr + i * 16] = f2bf(v.z);
    tile[tc + 3][tr + i * 16] = f2bf(v.w);
  }
  __syncthreads();
  const int wr = t >> 2;        // 0..63  (local n)
  const int wc = (t & 3) * 16;  // 0..48  (local k)
  const uint4 d0 = *(const uint4*)&tile[wr][wc];
  const uint4 d1 = *(const uint4*)&tile[wr][wc + 8];
  __bf16* dst = Wt + (size_t)(n0 + wr) * D_ + k0 + wc;
  *(uint4*)dst = d0;
  *(uint4*)(dst + 8) = d1;
}

// C = A[8192x1024] * Wt^T (Wt is bf16 [N][K]).  Block = 8 waves -> 128x128
// tile; each wave computes 32x64 (2x4 WMMA accumulators), K-loop step 32.
// ABF16:  0 = A is fp32 (converted on load), 1 = A is bf16.
// OUTMODE: 0 = bf16 [B][H][S][HD]  (Q, K)
//          1 = bf16 [B][H][HD][S]  (V transposed, contiguous PV operand)
//          2 = fp32 row-major [M][N] (final output)
template <int ABF16, int OUTMODE>
__global__ __launch_bounds__(256) void gemm_wmma(const void* __restrict__ Ap,
                                                 const __bf16* __restrict__ Wt,
                                                 void* __restrict__ Outp) {
  const int lane = threadIdx.x & 31;
  const int wave = threadIdx.x >> 5;
  const int m0 = blockIdx.x * 128 + (wave >> 1) * 32;
  const int n0 = blockIdx.y * 128 + (wave & 1) * 64;
  const int r0 = m0 + (lane & 15);
  const int c0 = n0 + (lane & 15);

  v8f acc[2][4];
#pragma unroll
  for (int i = 0; i < 2; ++i)
#pragma unroll
    for (int j = 0; j < 4; ++j) acc[i][j] = v8f_zero();

  for (int k = 0; k < D_; k += 32) {
    v16bf a0, a1;
    if (ABF16) {
      a0 = load_tile_bf16((const __bf16*)Ap, D_, r0,      k, lane);
      a1 = load_tile_bf16((const __bf16*)Ap, D_, r0 + 16, k, lane);
    } else {
      a0 = load_tile_f32((const float*)Ap, D_, r0,      k, lane);
      a1 = load_tile_f32((const float*)Ap, D_, r0 + 16, k, lane);
    }
#pragma unroll
    for (int j = 0; j < 4; ++j) {
      const v16bf bt = load_tile_bf16(Wt, D_, c0 + j * 16, k, lane);
      acc[0][j] = wmma_bf16(a0, bt, acc[0][j]);
      acc[1][j] = wmma_bf16(a1, bt, acc[1][j]);
    }
  }

  const int rowOff = ((lane >> 4) & 1) * 8;
#pragma unroll
  for (int mi = 0; mi < 2; ++mi)
#pragma unroll
    for (int j = 0; j < 4; ++j)
#pragma unroll
      for (int r = 0; r < 8; ++r) {
        const int m = m0 + mi * 16 + rowOff + r;
        const int n = n0 + j * 16 + (lane & 15);
        const float v = acc[mi][j][r];
        if (OUTMODE == 2) {
          ((float*)Outp)[(size_t)m * D_ + n] = v;
        } else {
          const int b = m >> 11, s = m & (S_ - 1);
          const int h = n >> 6,  hd = n & (HD_ - 1);
          __bf16* o = (__bf16*)Outp;
          if (OUTMODE == 0)
            o[(((size_t)(b * H_ + h)) * S_ + s) * HD_ + hd] = f2bf(v);
          else
            o[(((size_t)(b * H_ + h)) * HD_ + hd) * S_ + s] = f2bf(v);
        }
      }
}

// Flash attention: grid (S/128, B*H); 8 waves/block, each wave owns 16 query
// rows.  K/V stream in 32-key tiles, double-buffered in LDS via CDNA5 async
// global->LDS copies (ASYNCcnt): tile t+1 is staged while tile t computes, so
// each iteration pays one s_wait_asynccnt + one workgroup barrier.  Online
// softmax; keys past valid_len are skipped (exp(-1e6 - m) == 0 in fp32,
// identical to the reference mask semantics).
__global__ __launch_bounds__(256) void flash_attn(const __bf16* __restrict__ Q,
                                                  const __bf16* __restrict__ K,
                                                  const __bf16* __restrict__ Vt,
                                                  const int* __restrict__ valid_lens,
                                                  __bf16* __restrict__ O) {
  __shared__ __align__(16) __bf16 Kls[2][32][64];  // [buf][key local][hd]
  __shared__ __align__(16) __bf16 Vls[2][64][32];  // [buf][hd][key local]
  __shared__ __align__(16) __bf16 pls[8][16][32];  // per-wave P tile

  const int lane = threadIdx.x & 31;
  const int wave = threadIdx.x >> 5;
  const int bh = blockIdx.y;
  const int b = bh >> 4, h = bh & (H_ - 1);
  const int qBase = blockIdx.x * 128 + wave * 16;

  const __bf16* Qb = Q  + (size_t)bh * S_ * HD_;
  const __bf16* Kb = K  + (size_t)bh * S_ * HD_;
  const __bf16* Vb = Vt + (size_t)bh * HD_ * S_;

  const int vlen = valid_lens[b];
  const int nTiles = (vlen + 31) >> 5;  // block-uniform (per-batch)

  const v16bf aq0 = load_tile_bf16(Qb, HD_, qBase + (lane & 15), 0,  lane);
  const v16bf aq1 = load_tile_bf16(Qb, HD_, qBase + (lane & 15), 32, lane);

  // cooperative staging coordinates (per thread, loop-invariant)
  const int skr = threadIdx.x >> 3, skc = (threadIdx.x & 7) * 8;  // K: 32x64
  const int svr = threadIdx.x >> 2, svc = (threadIdx.x & 3) * 8;  // V: 64x32

  auto stage = [&](int tile, int buf) {
    const int kt = tile * 32;
    async_copy_b128(&Kls[buf][skr][skc], Kb + (size_t)(kt + skr) * HD_ + skc);
    async_copy_b128(&Vls[buf][svr][svc], Vb + (size_t)svr * S_ + kt + svc);
  };

  float mrow[8], lrow[8];
#pragma unroll
  for (int r = 0; r < 8; ++r) { mrow[r] = -3.0e38f; lrow[r] = 0.0f; }
  v8f o0 = v8f_zero(), o1 = v8f_zero(), o2 = v8f_zero(), o3 = v8f_zero();

  stage(0, 0);  // prologue: tile 0 into buffer 0

  for (int t = 0; t < nTiles; ++t) {
    const int buf = t & 1;
    // my async writes for `buf` have landed; barrier makes everyone's visible
    // and guarantees all readers of `buf^1` (iteration t-1) are done.
    wait_async0();
    __syncthreads();
    if (t + 1 < nTiles) stage(t + 1, buf ^ 1);

    const __bf16* Kt = &Kls[buf][0][0];
    const __bf16* Vv = &Vls[buf][0][0];
    const int kt = t * 32;

    // S = Q * K^T  (16 queries x 32 keys), fp32 accum, scale 1/sqrt(64)
    v8f c0 = v8f_zero(), c1 = v8f_zero();
    c0 = wmma_bf16(aq0, load_tile_bf16(Kt, 64,      (lane & 15), 0,  lane), c0);
    c0 = wmma_bf16(aq1, load_tile_bf16(Kt, 64,      (lane & 15), 32, lane), c0);
    c1 = wmma_bf16(aq0, load_tile_bf16(Kt, 64, 16 + (lane & 15), 0,  lane), c1);
    c1 = wmma_bf16(aq1, load_tile_bf16(Kt, 64, 16 + (lane & 15), 32, lane), c1);

    const bool ok0 = (kt      + (lane & 15)) < vlen;
    const bool ok1 = (kt + 16 + (lane & 15)) < vlen;
    float mt[8];
#pragma unroll
    for (int r = 0; r < 8; ++r) {
      c0[r] = ok0 ? c0[r] * 0.125f : -1.0e6f;
      c1[r] = ok1 ? c1[r] * 0.125f : -1.0e6f;
      mt[r] = fmaxf(c0[r], c1[r]);
    }
    // row reduction across the 16 lanes holding one matrix row
#pragma unroll
    for (int off = 1; off < 16; off <<= 1)
#pragma unroll
      for (int r = 0; r < 8; ++r)
        mt[r] = fmaxf(mt[r], __shfl_xor(mt[r], off, 32));

    float alpha[8];
#pragma unroll
    for (int r = 0; r < 8; ++r) {
      const float mnew = fmaxf(mrow[r], mt[r]);
      alpha[r] = __expf(mrow[r] - mnew);
      mrow[r] = mnew;
    }

    const int prow = ((lane >> 4) & 1) * 8;
    float ps[8];
#pragma unroll
    for (int r = 0; r < 8; ++r) {
      const float p0 = __expf(c0[r] - mrow[r]);
      const float p1 = __expf(c1[r] - mrow[r]);
      pls[wave][prow + r][lane & 15]        = f2bf(p0);
      pls[wave][prow + r][16 + (lane & 15)] = f2bf(p1);
      ps[r] = p0 + p1;
    }
#pragma unroll
    for (int off = 1; off < 16; off <<= 1)
#pragma unroll
      for (int r = 0; r < 8; ++r)
        ps[r] += __shfl_xor(ps[r], off, 32);
#pragma unroll
    for (int r = 0; r < 8; ++r) lrow[r] = lrow[r] * alpha[r] + ps[r];

#pragma unroll
    for (int r = 0; r < 8; ++r) {
      o0[r] *= alpha[r]; o1[r] *= alpha[r];
      o2[r] *= alpha[r]; o3[r] *= alpha[r];
    }

    // P (16x32, A layout) from per-wave LDS; V tiles from staged LDS
    const v16bf ap = load_tile_bf16(&pls[wave][0][0], 32, lane & 15, 0, lane);
    o0 = wmma_bf16(ap, load_tile_bf16(Vv, 32,      (lane & 15), 0, lane), o0);
    o1 = wmma_bf16(ap, load_tile_bf16(Vv, 32, 16 + (lane & 15), 0, lane), o1);
    o2 = wmma_bf16(ap, load_tile_bf16(Vv, 32, 32 + (lane & 15), 0, lane), o2);
    o3 = wmma_bf16(ap, load_tile_bf16(Vv, 32, 48 + (lane & 15), 0, lane), o3);
  }

  // normalize and merge heads: out bf16 [B*S][D]
  const int rowOff = ((lane >> 4) & 1) * 8;
#pragma unroll
  for (int r = 0; r < 8; ++r) {
    const float inv = 1.0f / lrow[r];
    const int q = qBase + rowOff + r;
    const size_t base = ((size_t)b * S_ + q) * D_ + h * HD_;
    O[base +      (lane & 15)] = f2bf(o0[r] * inv);
    O[base + 16 + (lane & 15)] = f2bf(o1[r] * inv);
    O[base + 32 + (lane & 15)] = f2bf(o2[r] * inv);
    O[base + 48 + (lane & 15)] = f2bf(o3[r] * inv);
  }
}

extern "C" void kernel_launch(void* const* d_in, const int* in_sizes, int n_in,
                              void* d_out, int out_size, void* d_ws, size_t ws_size,
                              hipStream_t stream) {
  const float* queries    = (const float*)d_in[0];
  const float* keys       = (const float*)d_in[1];
  const float* values     = (const float*)d_in[2];
  const int*   valid_lens = (const int*)d_in[3];
  const float* Wq = (const float*)d_in[4];
  const float* Wk = (const float*)d_in[5];
  const float* Wv = (const float*)d_in[6];
  const float* Wo = (const float*)d_in[7];

  const size_t nQKV = (size_t)B_ * H_ * S_ * HD_;  // 8M bf16 elems
  const size_t nW   = (size_t)D_ * D_;             // 1M bf16 elems
  __bf16* q_ws = (__bf16*)d_ws;
  __bf16* k_ws = q_ws + nQKV;
  __bf16* v_ws = k_ws + nQKV;        // stored [bh][hd][s]
  __bf16* a_ws = v_ws + nQKV;        // attention out, bf16 [B*S][D]
  // transposed Wq/Wk/Wv alias the a_ws region: they are fully consumed by the
  // projection GEMMs before flash_attn writes a_ws.  Wo^T lives past a_ws.
  __bf16* wq_t = a_ws;
  __bf16* wk_t = wq_t + nW;
  __bf16* wv_t = wk_t + nW;
  __bf16* wo_t = a_ws + nQKV;

  dim3 blk(256);
  dim3 tgrid(D_ / 64, D_ / 64);  // 16 x 16
  transpose_w_bf16<<<tgrid, blk, 0, stream>>>(Wq, wq_t);
  transpose_w_bf16<<<tgrid, blk, 0, stream>>>(Wk, wk_t);
  transpose_w_bf16<<<tgrid, blk, 0, stream>>>(Wv, wv_t);
  transpose_w_bf16<<<tgrid, blk, 0, stream>>>(Wo, wo_t);

  dim3 ggrid(B_ * S_ / 128, D_ / 128);  // 64 x 8
  gemm_wmma<0, 0><<<ggrid, blk, 0, stream>>>(queries, wq_t, q_ws);
  gemm_wmma<0, 0><<<ggrid, blk, 0, stream>>>(keys,    wk_t, k_ws);
  gemm_wmma<0, 1><<<ggrid, blk, 0, stream>>>(values,  wv_t, v_ws);

  dim3 agrid(S_ / 128, B_ * H_);        // 16 x 64
  flash_attn<<<agrid, blk, 0, stream>>>(q_ws, k_ws, v_ws, valid_lens, a_ws);

  gemm_wmma<1, 2><<<ggrid, blk, 0, stream>>>(a_ws, wo_t, (float*)d_out);
}
